// Seq2Seq_1992864825585
// MI455X (gfx1250) — matrix operand order repaired and checked by reading, not compiled, not run on hardware
//
#include <hip/hip_runtime.h>
#include <hip/hip_bf16.h>
#include <math.h>

typedef __bf16 bf16_t;
typedef __attribute__((ext_vector_type(16))) __bf16          v16bf;
typedef __attribute__((ext_vector_type(8)))  float           v8f;
typedef __attribute__((ext_vector_type(8)))  unsigned short  v8us;
typedef __attribute__((ext_vector_type(4)))  unsigned int    v4u;
typedef __attribute__((ext_vector_type(8)))  int             v8i;
typedef __attribute__((ext_vector_type(4)))  int             v4i;

#define VOCAB 32000
#define HDIM  1024
#define LNUM  2
#define BATCH 64
#define SLEN  128
#define TLEN  32

// ---------------- helpers ----------------

__device__ inline bf16_t f2bf(float f) {
    unsigned u = __builtin_bit_cast(unsigned, f);
    u += 0x7FFFu + ((u >> 16) & 1u);                 // round-to-nearest-even
    unsigned short h = (unsigned short)(u >> 16);
    return __builtin_bit_cast(bf16_t, h);
}

__device__ inline float sigmoidf_(float x) { return 1.0f / (1.0f + __expf(-x)); }

__device__ inline v8f wmma_bf16(v16bf a, v16bf b, v8f c) {
    return __builtin_amdgcn_wmma_f32_16x16x32_bf16(false, a, false, b, (short)0, c,
                                                   false, false);
}

// A fragment: 16x32 bf16 tile staged row-major in LDS (ld = 32).
// lane 0-15: row M=lane, K halves {kb..kb+7} and {16+kb..16+kb+7}, kb = 8*(lane>=16)
__device__ inline v16bf load_a_frag(const bf16_t* lds, int m0, int lane) {
    const int row = m0 + (lane & 15);
    const int kb  = (lane >> 4) << 3;
    const bf16_t* p = lds + row * 32;
    union { v16bf v; v8us u[2]; } t;
    t.u[0] = *(const v8us*)(p + kb);
    t.u[1] = *(const v8us*)(p + 16 + kb);
    return t.v;
}

// B fragment for D = A x B from row-major weight W[rows][HDIM]; B[k][n] = W[n][k].
// lane = column n0+(lane&15); K halves {kb..kb+7} and {16+kb..16+kb+7}, kb = k0 + 8*(lane>=16)
__device__ inline v16bf load_b_frag(const bf16_t* __restrict__ W, int n0, int k0, int lane) {
    const int col = n0 + (lane & 15);
    const int kb  = k0 + ((lane >> 4) << 3);
    const bf16_t* p = W + (size_t)col * HDIM + kb;
    union { v16bf v; v8us u[2]; } t;
    t.u[0] = *(const v8us*)(p);
    t.u[1] = *(const v8us*)(p + 16);
    return t.v;
}

// ---- TDM: load a 64-row x 32-col bf16 tile (row stride HDIM elems) into LDS ----
// D# group0: count=1, lds_addr, global_addr[56:0], type=2 (bits 127:126).
// D# group1: data_size=1 (2B), tensor_dim0=1024, tensor_dim1=64, tile_dim0=32,
//            tile_dim1=64, tensor_dim0_stride=1024. Groups 2/3/4 unused (2D tile).
__device__ inline void tdm_load_tile(unsigned lds_off, const bf16_t* gptr) {
    unsigned long long ga = (unsigned long long)gptr;
    v4u g0;
    g0[0] = 1u;                                            // count=1
    g0[1] = lds_off;                                       // lds_addr (bytes)
    g0[2] = (unsigned)ga;                                  // global_addr[31:0]
    g0[3] = (unsigned)((ga >> 32) & 0x01FFFFFFu) | (2u << 30); // addr[56:32] | type=2
    v8i g1;
    g1[0] = (int)(1u << 16);                               // data_size=1 -> 2 bytes
    g1[1] = (int)((1024u & 0xFFFFu) << 16);                // tensor_dim0 lo16 @bits63:48
    g1[2] = (int)(((1024u >> 16) & 0xFFFFu) | (64u << 16));// dim0 hi16 | tensor_dim1 lo16
    g1[3] = (int)(32u << 16);                              // dim1 hi16=0 | tile_dim0=32
    g1[4] = (int)64u;                                      // tile_dim1=64, tile_dim2=0
    g1[5] = (int)1024;                                     // tensor_dim0_stride[31:0]
    g1[6] = 0;                                             // stride hi | dim1_stride lo
    g1[7] = 0;
    v4i g2 = {0, 0, 0, 0};
    v4i g3 = {0, 0, 0, 0};
    v8i g4 = {0, 0, 0, 0, 0, 0, 0, 0};
    __builtin_amdgcn_tensor_load_to_lds(g0, g1, g2, g3, g4, 0);
}

// ---------------- elementwise kernels ----------------

__global__ void convert_kernel(const float* __restrict__ src, bf16_t* __restrict__ dst,
                               size_t n) {
    size_t i = (size_t)blockIdx.x * blockDim.x + threadIdx.x;
    size_t st = (size_t)gridDim.x * blockDim.x;
    for (; i < n; i += st) dst[i] = f2bf(src[i]);
}

__global__ void zero_f32_kernel(float* __restrict__ p, size_t n) {
    size_t i = (size_t)blockIdx.x * blockDim.x + threadIdx.x;
    size_t st = (size_t)gridDim.x * blockDim.x;
    for (; i < n; i += st) p[i] = 0.0f;
}

__global__ void zero_bf_kernel(bf16_t* __restrict__ p, size_t n) {
    size_t i = (size_t)blockIdx.x * blockDim.x + threadIdx.x;
    size_t st = (size_t)gridDim.x * blockDim.x;
    for (; i < n; i += st) p[i] = __builtin_bit_cast(bf16_t, (unsigned short)0);
}

// outputs[:,0,:] = 0
__global__ void zero_out0_kernel(float* __restrict__ out) {
    float* p = out + (size_t)blockIdx.x * TLEN * VOCAB;
    for (int v = threadIdx.x; v < VOCAB; v += blockDim.x) p[v] = 0.0f;
}

// x_bf[b,:] = bf16(emb[tokbase[b*stride], :])
__global__ void gather_kernel(const float* __restrict__ emb, const int* __restrict__ tokbase,
                              int stride, bf16_t* __restrict__ xout) {
    const int b = blockIdx.x;
    const int tokid = tokbase[(size_t)b * stride];
    const float* src = emb + (size_t)tokid * HDIM;
    for (int e = threadIdx.x; e < HDIM; e += blockDim.x)
        xout[(size_t)b * HDIM + e] = f2bf(src[e]);
}

// ---------------- fused GRU layer ----------------
// grid = HDIM/32 blocks, 256 threads (8 waves). Block owns output columns [jb, jb+32).
// wave -> M subtile m (0..3), N subtile nn (0..1): one 16x16 subtile, 6 accumulators
// (gi/gh x r,z,n). A tiles (x, h) double-buffered in LDS via TDM, prefetched one
// K-iteration ahead by wave 0 (TENSORcnt pipelining).

__global__ __launch_bounds__(256)
void gru_layer_kernel(const bf16_t* __restrict__ x_bf,     // [64,1024]
                      const bf16_t* __restrict__ hprev_bf, // [64,1024]
                      const float*  __restrict__ hprev_f32,// [64,1024]
                      const bf16_t* __restrict__ Wih,      // [3072,1024]
                      const bf16_t* __restrict__ Whh,      // [3072,1024]
                      const float*  __restrict__ bih,      // [3072]
                      const float*  __restrict__ bhh,      // [3072]
                      float*        __restrict__ hnew_f32, // [64,1024]
                      bf16_t*       __restrict__ hnew_bf,  // [64,1024]
                      const int*    __restrict__ src_len,  // [64] or unused
                      int use_mask, int step)
{
    __shared__ bf16_t ldsX[2][BATCH * 32];
    __shared__ bf16_t ldsH[2][BATCH * 32];

    const int tid  = threadIdx.x;
    const int lane = tid & 31;
    const int wave = tid >> 5;
    const int m    = wave & 3;
    const int nn   = wave >> 2;
    const int jb   = blockIdx.x * 32;
    const int n0   = jb + nn * 16;

    const unsigned ldsXo[2] = { (unsigned)(unsigned long long)&ldsX[0][0],
                                (unsigned)(unsigned long long)&ldsX[1][0] };
    const unsigned ldsHo[2] = { (unsigned)(unsigned long long)&ldsH[0][0],
                                (unsigned)(unsigned long long)&ldsH[1][0] };

    v8f zz = {};
    v8f acc_i[3] = { zz, zz, zz };
    v8f acc_h[3] = { zz, zz, zz };

    if (wave == 0) {                      // preload K-iter 0
        tdm_load_tile(ldsXo[0], x_bf);
        tdm_load_tile(ldsHo[0], hprev_bf);
    }

    for (int it = 0; it < HDIM / 32; ++it) {
        const int k0 = it * 32;
        if (wave == 0) {
            if (it < HDIM / 32 - 1) {     // prefetch next tile pair
                tdm_load_tile(ldsXo[(it + 1) & 1], x_bf + k0 + 32);
                tdm_load_tile(ldsHo[(it + 1) & 1], hprev_bf + k0 + 32);
                __builtin_amdgcn_s_wait_tensorcnt((short)2);  // current pair done
            } else {
                __builtin_amdgcn_s_wait_tensorcnt((short)0);
            }
        }
        __syncthreads();                  // tiles for iter `it` visible to all waves

        const bf16_t* lx = &ldsX[it & 1][0];
        const bf16_t* lh = &ldsH[it & 1][0];
        v16bf ax = load_a_frag(lx, m * 16, lane);
        v16bf ah = load_a_frag(lh, m * 16, lane);
#pragma unroll
        for (int p = 0; p < 3; ++p) {
            v16bf bi = load_b_frag(Wih, n0 + p * HDIM, k0, lane);
            acc_i[p] = wmma_bf16(ax, bi, acc_i[p]);
            v16bf bh = load_b_frag(Whh, n0 + p * HDIM, k0, lane);
            acc_h[p] = wmma_bf16(ah, bh, acc_h[p]);
        }
        __syncthreads();                  // reads done before buffer is overwritten
    }

    const int rbase = m * 16 + ((lane >> 4) << 3);
    const int col   = n0 + (lane & 15);
    const float bir = bih[col];            const float bhr = bhh[col];
    const float biz = bih[HDIM + col];     const float bhz = bhh[HDIM + col];
    const float bin = bih[2 * HDIM + col]; const float bhn = bhh[2 * HDIM + col];
#pragma unroll
    for (int r = 0; r < 8; ++r) {
        const int row = rbase + r;        // batch index
        const float rg = sigmoidf_(acc_i[0][r] + bir + acc_h[0][r] + bhr);
        const float zg = sigmoidf_(acc_i[1][r] + biz + acc_h[1][r] + bhz);
        const float ng = tanhf(acc_i[2][r] + bin + rg * (acc_h[2][r] + bhn));
        const float hp = hprev_f32[(size_t)row * HDIM + col];
        float hn = (1.0f - zg) * ng + zg * hp;
        if (use_mask && !(step < src_len[row])) hn = hp;
        hnew_f32[(size_t)row * HDIM + col] = hn;
        hnew_bf[(size_t)row * HDIM + col]  = f2bf(hn);
    }
}

// ---------------- fc: pred = h1 @ fcW^T + b, written to out[:, t, :] ----------------
// grid = VOCAB/64 blocks. A tile staged with GLOBAL_LOAD_ASYNC_TO_LDS (ASYNCcnt).

__global__ __launch_bounds__(256)
void fc_kernel(const bf16_t* __restrict__ h_bf,  // [64,1024]
               const bf16_t* __restrict__ Wf,    // [32000,1024]
               const float*  __restrict__ bias,  // [32000]
               float*        __restrict__ out,   // [64,32,32000]
               int t)
{
    __shared__ bf16_t ldsA[BATCH * 32];

    const int tid  = threadIdx.x;
    const int lane = tid & 31;
    const int wave = tid >> 5;
    const int m    = wave & 3;
    const int np   = wave >> 2;
    const int nb   = blockIdx.x * 64;

    v8f zz = {};
    v8f acc[2] = { zz, zz };

    const int rr = tid >> 2;              // row 0..63
    const int cc = (tid & 3) * 8;         // col chunk 0/8/16/24
    const unsigned ldso =
        (unsigned)(unsigned long long)&ldsA[0] + (unsigned)((rr * 32 + cc) * 2);

    for (int k0 = 0; k0 < HDIM; k0 += 32) {
        unsigned long long ga =
            (unsigned long long)(h_bf + (size_t)rr * HDIM + k0 + cc);
        asm volatile("global_load_async_to_lds_b128 %0, %1, off"
                     :: "v"(ldso), "v"(ga) : "memory");
        asm volatile("s_wait_asynccnt 0x0" ::: "memory");
        __syncthreads();

        v16bf a = load_a_frag(ldsA, m * 16, lane);
#pragma unroll
        for (int s = 0; s < 2; ++s) {
            v16bf b = load_b_frag(Wf, nb + (np * 2 + s) * 16, k0, lane);
            acc[s] = wmma_bf16(a, b, acc[s]);
        }
        __syncthreads();
    }

    const int rbase = m * 16 + ((lane >> 4) << 3);
#pragma unroll
    for (int s = 0; s < 2; ++s) {
        const int col = nb + (np * 2 + s) * 16 + (lane & 15);
        const float bv = bias[col];
#pragma unroll
        for (int r = 0; r < 8; ++r) {
            const int row = rbase + r;
            out[(size_t)row * TLEN * VOCAB + (size_t)t * VOCAB + col] = acc[s][r] + bv;
        }
    }
}

// ---------------- argmax over V per batch row (first-max tie rule) ----------------

__global__ __launch_bounds__(256)
void argmax_kernel(const float* __restrict__ out, int t, int* __restrict__ tok) {
    const int b = blockIdx.x;
    const int tid = threadIdx.x;
    const float* p = out + (size_t)b * TLEN * VOCAB + (size_t)t * VOCAB;
    float best = -__builtin_inff();
    int bidx = 0;
    for (int v = tid; v < VOCAB; v += 256) {
        const float val = p[v];
        if (val > best) { best = val; bidx = v; }   // ascending scan keeps lowest idx
    }
    __shared__ float sv[256];
    __shared__ int   si[256];
    sv[tid] = best; si[tid] = bidx;
    __syncthreads();
    for (int ofs = 128; ofs > 0; ofs >>= 1) {
        if (tid < ofs) {
            if (sv[tid + ofs] > sv[tid] ||
                (sv[tid + ofs] == sv[tid] && si[tid + ofs] < si[tid])) {
                sv[tid] = sv[tid + ofs];
                si[tid] = si[tid + ofs];
            }
        }
        __syncthreads();
    }
    if (tid == 0) tok[b] = si[0];
}

// ---------------- host orchestration ----------------

extern "C" void kernel_launch(void* const* d_in, const int* in_sizes, int n_in,
                              void* d_out, int out_size, void* d_ws, size_t ws_size,
                              hipStream_t stream)
{
    const int*   src     = (const int*)  d_in[0];
    const int*   src_len = (const int*)  d_in[1];
    const int*   trg     = (const int*)  d_in[2];
    const float* emb     = (const float*)d_in[3];
    const float* encWih  = (const float*)d_in[4];
    const float* encWhh  = (const float*)d_in[5];
    const float* encBih  = (const float*)d_in[6];
    const float* encBhh  = (const float*)d_in[7];
    const float* decWih  = (const float*)d_in[8];
    const float* decWhh  = (const float*)d_in[9];
    const float* decBih  = (const float*)d_in[10];
    const float* decBhh  = (const float*)d_in[11];
    const float* fcW     = (const float*)d_in[12];
    const float* fcB     = (const float*)d_in[13];
    float* out = (float*)d_out;

    // ---- workspace carve (256B aligned) ----
    char* ws = (char*)d_ws;
    size_t off = 0;
    auto carve = [&](size_t bytes) -> char* {
        size_t o = (off + 255) & ~(size_t)255;
        off = o + bytes;
        return ws + o;
    };
    const size_t GRUW   = (size_t)LNUM * 3 * HDIM * HDIM;  // elements per W tensor
    const size_t LSTRID = (size_t)3 * HDIM * HDIM;         // per-layer element stride

    bf16_t* encWih_bf = (bf16_t*)carve(GRUW * 2);
    bf16_t* encWhh_bf = (bf16_t*)carve(GRUW * 2);
    bf16_t* decWih_bf = (bf16_t*)carve(GRUW * 2);
    bf16_t* decWhh_bf = (bf16_t*)carve(GRUW * 2);
    bf16_t* fcW_bf    = (bf16_t*)carve((size_t)VOCAB * HDIM * 2);
    float*  h_f32     = (float*) carve((size_t)2 * LNUM * BATCH * HDIM * 4);
    bf16_t* h_bf      = (bf16_t*)carve((size_t)2 * LNUM * BATCH * HDIM * 2);
    bf16_t* x_bf      = (bf16_t*)carve((size_t)BATCH * HDIM * 2);
    int*    tok       = (int*)   carve((size_t)BATCH * 4);
    if (off > ws_size) return;  // not enough scratch

    auto hF = [&](int buf, int layer) {
        return h_f32 + ((size_t)buf * LNUM + layer) * BATCH * HDIM;
    };
    auto hB = [&](int buf, int layer) {
        return h_bf + ((size_t)buf * LNUM + layer) * BATCH * HDIM;
    };

    // ---- per-call weight conversion fp32 -> bf16 (L2-resident afterwards) ----
    convert_kernel<<<2048, 256, 0, stream>>>(encWih, encWih_bf, GRUW);
    convert_kernel<<<2048, 256, 0, stream>>>(encWhh, encWhh_bf, GRUW);
    convert_kernel<<<2048, 256, 0, stream>>>(decWih, decWih_bf, GRUW);
    convert_kernel<<<2048, 256, 0, stream>>>(decWhh, decWhh_bf, GRUW);
    convert_kernel<<<4096, 256, 0, stream>>>(fcW, fcW_bf, (size_t)VOCAB * HDIM);

    // ---- init state + outputs[:,0,:] ----
    zero_f32_kernel<<<512, 256, 0, stream>>>(h_f32, (size_t)2 * LNUM * BATCH * HDIM);
    zero_bf_kernel <<<512, 256, 0, stream>>>(h_bf,  (size_t)2 * LNUM * BATCH * HDIM);
    zero_out0_kernel<<<BATCH, 256, 0, stream>>>(out);

    int par = 0;

    // ---- encoder: S masked steps x 2 layers ----
    for (int t = 0; t < SLEN; ++t) {
        gather_kernel<<<BATCH, 256, 0, stream>>>(emb, src + t, SLEN, x_bf);
        gru_layer_kernel<<<HDIM / 32, 256, 0, stream>>>(
            x_bf, hB(par, 0), hF(par, 0),
            encWih_bf, encWhh_bf, encBih, encBhh,
            hF(par ^ 1, 0), hB(par ^ 1, 0), src_len, 1, t);
        gru_layer_kernel<<<HDIM / 32, 256, 0, stream>>>(
            hB(par ^ 1, 0), hB(par, 1), hF(par, 1),
            encWih_bf + LSTRID, encWhh_bf + LSTRID, encBih + 3 * HDIM, encBhh + 3 * HDIM,
            hF(par ^ 1, 1), hB(par ^ 1, 1), src_len, 1, t);
        par ^= 1;
    }

    // ---- decoder: T-1 greedy autoregressive steps ----
    for (int d = 0; d < TLEN - 1; ++d) {
        if (d == 0)
            gather_kernel<<<BATCH, 256, 0, stream>>>(emb, trg, TLEN, x_bf); // trg[:,0]
        else
            gather_kernel<<<BATCH, 256, 0, stream>>>(emb, tok, 1, x_bf);
        gru_layer_kernel<<<HDIM / 32, 256, 0, stream>>>(
            x_bf, hB(par, 0), hF(par, 0),
            decWih_bf, decWhh_bf, decBih, decBhh,
            hF(par ^ 1, 0), hB(par ^ 1, 0), (const int*)nullptr, 0, 0);
        gru_layer_kernel<<<HDIM / 32, 256, 0, stream>>>(
            hB(par ^ 1, 0), hB(par, 1), hF(par, 1),
            decWih_bf + LSTRID, decWhh_bf + LSTRID, decBih + 3 * HDIM, decBhh + 3 * HDIM,
            hF(par ^ 1, 1), hB(par ^ 1, 1), (const int*)nullptr, 0, 0);
        fc_kernel<<<VOCAB / 64, 256, 0, stream>>>(hB(par ^ 1, 1), fcW_bf, fcB, out, d + 1);
        if (d < TLEN - 2)
            argmax_kernel<<<BATCH, 256, 0, stream>>>(out, d + 1, tok);
        par ^= 1;
    }
}
